// ScaledDotProductAttention_11982958755996
// MI455X (gfx1250) — compile-verified, matrix operand order checked by reading, not run on hardware
//
#include <hip/hip_runtime.h>
#include <math.h>

// ---------------------------------------------------------------------------
// Flash-attention forward for B=1, H=16, S=4096, DK=64 (fp32 in/out),
// causal mask, computed with v_wmma_f32_16x16x32_f16 on gfx1250 (wave32).
// Softmax row reductions via DPP16 butterflies; softmax runs in the base-2
// domain (log2e folded into the Q scale) so exp == one v_exp_f32.
// ---------------------------------------------------------------------------

typedef _Float16 v16h __attribute__((ext_vector_type(16)));
typedef _Float16 h8   __attribute__((ext_vector_type(8)));
typedef float    v8f  __attribute__((ext_vector_type(8)));

#define HEADS 16
#define SEQ   4096
#define DKDIM 64
#define BM    64      // query rows per workgroup
#define BN    64      // keys per iteration
#define NWAVE 4       // 4 wave32 -> 128 threads

// 1/sqrt(64) * log2(e): run softmax in base-2 (v_exp_f32 is exp2 natively).
#define QSCALE 0.18033688011112042591999058512524f

#if __has_builtin(__builtin_amdgcn_exp2f)
#define EXP2F(x) __builtin_amdgcn_exp2f(x)
#else
#define EXP2F(x) exp2f(x)
#endif

// v16h A/B operand assembled from two contiguous 16-byte LDS reads.
// Elements 0..7 live in VGPRs 0..3 (K = kbase+0..7), elements 8..15 in
// VGPRs 4..7 (K = 16+kbase+0..7) -- matches the 16-bit 16x32 operand layout.
union V16U {
    v16h v;
    h8   h[2];
};

// DPP16 lane permute within a 16-lane row (VALU only, no LDS).
template <int CTRL>
__device__ __forceinline__ float dpp_perm(float x) {
    union { float f; int i; } u, r;
    u.f = x;
    r.i = __builtin_amdgcn_update_dpp(u.i, u.i, CTRL, 0xF, 0xF, true);
    return r.f;
}

// Full 16-lane butterfly reductions: xor1, xor2, xor7(half-mirror), xor15(mirror).
__device__ __forceinline__ float row_max16(float x) {
    x = fmaxf(x, dpp_perm<0x0B1>(x));   // quad_perm [1,0,3,2]
    x = fmaxf(x, dpp_perm<0x04E>(x));   // quad_perm [2,3,0,1]
    x = fmaxf(x, dpp_perm<0x141>(x));   // row_half_mirror
    x = fmaxf(x, dpp_perm<0x140>(x));   // row_mirror
    return x;
}
__device__ __forceinline__ float row_sum16(float x) {
    x += dpp_perm<0x0B1>(x);
    x += dpp_perm<0x04E>(x);
    x += dpp_perm<0x141>(x);
    x += dpp_perm<0x140>(x);
    return x;
}

__launch_bounds__(128)
__global__ void fa_fwd_wmma_kernel(const float* __restrict__ Q,
                                   const float* __restrict__ K,
                                   const float* __restrict__ V,
                                   float* __restrict__ O) {
    __shared__ __align__(16) _Float16 sK [BN][DKDIM];       // K block, f16, row-major
    __shared__ __align__(16) _Float16 sVT[DKDIM][BN];       // V block, f16, transposed
    __shared__ __align__(16) _Float16 sP [NWAVE][16][BN];   // per-wave P staging

    const int tid  = threadIdx.x;
    const int wave = tid >> 5;
    const int lane = tid & 31;
    const int half = lane >> 4;   // 0: lanes 0-15, 1: lanes 16-31
    const int lp   = lane & 15;
    const int kb   = half * 8;    // kbase for 16-bit operand layout

    const int ib = blockIdx.x;                       // query block index
    const int h  = blockIdx.y;                       // head
    const size_t hoff = (size_t)h * SEQ * DKDIM;
    const int m0 = ib * BM + wave * 16;              // first query row of this wave

    // ---- load Q A-operands once (0.125*log2e folded into the cvt) ----
    V16U aQ0, aQ1;
    {
        const float* qr = Q + hoff + (size_t)(m0 + lp) * DKDIM;
#pragma unroll
        for (int e = 0; e < 8; ++e) {
            aQ0.v[e]     = (_Float16)(qr[kb + e]      * QSCALE);
            aQ0.v[8 + e] = (_Float16)(qr[16 + kb + e] * QSCALE);
            aQ1.v[e]     = (_Float16)(qr[32 + kb + e] * QSCALE);
            aQ1.v[8 + e] = (_Float16)(qr[48 + kb + e] * QSCALE);
        }
    }

    v8f acc[4] = {v8f{}, v8f{}, v8f{}, v8f{}};       // 16x64 f32 output tile
    float rowmax[8], rowsum[8];
#pragma unroll
    for (int r = 0; r < 8; ++r) { rowmax[r] = -INFINITY; rowsum[r] = 0.0f; }

    // staging roles (128 threads)
    const int krow  = tid >> 1;          // 0..63 : K row inside block
    const int kcol0 = (tid & 1) * 32;    // 32-dim half
    const int vdim  = tid >> 1;          // 0..63 : V dim (column of V)
    const int vk0   = (tid & 1) * 32;    // 32-key half

    const int nblk = ib + 1;             // causal: only key blocks <= ib
    for (int j = 0; j < nblk; ++j) {
        __syncthreads();                 // previous iteration done reading LDS

        // ---- stage K block: f32 -> f16, row-major ----
        {
            const float* kr = K + hoff + (size_t)(j * BN + krow) * DKDIM + kcol0;
#pragma unroll
            for (int c = 0; c < 32; c += 8) {
                h8 t;
#pragma unroll
                for (int e = 0; e < 8; ++e) t[e] = (_Float16)kr[c + e];
                *(h8*)&sK[krow][kcol0 + c] = t;
            }
        }
        // ---- stage V block: f32 -> f16, transposed (sVT[dim][key]) ----
        {
            const float* vb = V + hoff + (size_t)(j * BN + vk0) * DKDIM + vdim;
#pragma unroll
            for (int kk = 0; kk < 32; kk += 8) {
                h8 t;
#pragma unroll
                for (int e = 0; e < 8; ++e) t[e] = (_Float16)vb[(size_t)(kk + e) * DKDIM];
                *(h8*)&sVT[vdim][vk0 + kk] = t;
            }
        }
        // ---- prefetch next K/V block toward L2 (global_prefetch_b8) ----
        if (j + 1 < nblk) {
            __builtin_prefetch(K + hoff + (size_t)((j + 1) * BN + krow) * DKDIM + kcol0, 0, 1);
            __builtin_prefetch(V + hoff + (size_t)((j + 1) * BN + vk0) * DKDIM + vdim, 0, 1);
        }
        __syncthreads();

        // ---- S = Q K^T : 4 column subtiles, 2 chained WMMAs (K=64) each ----
        v8f sc[4];
#pragma unroll
        for (int nt = 0; nt < 4; ++nt) {
            const _Float16* kp = &sK[nt * 16 + lp][0];   // B[k][n]=K[n][k] -> row read
            V16U b0, b1;
            b0.h[0] = *(const h8*)(kp + kb);
            b0.h[1] = *(const h8*)(kp + 16 + kb);
            b1.h[0] = *(const h8*)(kp + 32 + kb);
            b1.h[1] = *(const h8*)(kp + 48 + kb);
            v8f c = {};
            c = __builtin_amdgcn_wmma_f32_16x16x32_f16(false, aQ0.v, false, b0.v,
                                                       (short)0, c, false, false);
            c = __builtin_amdgcn_wmma_f32_16x16x32_f16(false, aQ1.v, false, b1.v,
                                                       (short)0, c, false, false);
            sc[nt] = c;
        }

        // ---- causal mask: only the diagonal key block needs it ----
        if (j == ib) {
#pragma unroll
            for (int nt = 0; nt < 4; ++nt) {
                const int gcol = j * BN + nt * 16 + lp;
#pragma unroll
                for (int r = 0; r < 8; ++r) {
                    const int grow = m0 + half * 8 + r;
                    if (gcol > grow) sc[nt][r] = -INFINITY;
                }
            }
        }

        // ---- online softmax (base-2 domain): rows striped across 8 C VGPRs,
        //      columns across a 16-lane DPP row -> VALU-only butterflies ----
        float mnew[8];
#pragma unroll
        for (int r = 0; r < 8; ++r) {
            float m = fmaxf(fmaxf(sc[0][r], sc[1][r]), fmaxf(sc[2][r], sc[3][r]));
            mnew[r] = row_max16(m);
        }

        float scl[8], psum[8];
#pragma unroll
        for (int r = 0; r < 8; ++r) {
            mnew[r]   = fmaxf(mnew[r], rowmax[r]);
            scl[r]    = EXP2F(rowmax[r] - mnew[r]);
            rowmax[r] = mnew[r];
            psum[r]   = 0.0f;
        }

        // P = exp2(S - mnew); stage row-major into per-wave LDS tile
#pragma unroll
        for (int nt = 0; nt < 4; ++nt) {
#pragma unroll
            for (int r = 0; r < 8; ++r) {
                const float p = EXP2F(sc[nt][r] - mnew[r]);
                psum[r] += p;
                sP[wave][half * 8 + r][nt * 16 + lp] = (_Float16)p;
            }
        }
#pragma unroll
        for (int r = 0; r < 8; ++r) {
            rowsum[r] = rowsum[r] * scl[r] + row_sum16(psum[r]);
            acc[0][r] *= scl[r];
            acc[1][r] *= scl[r];
            acc[2][r] *= scl[r];
            acc[3][r] *= scl[r];
        }

        // wave-private LDS tile: wait for our own ds stores before re-reading
        asm volatile("s_wait_dscnt 0" ::: "memory");

        // ---- O += P V : A = P (16x32 slices), B = V^T rows (contiguous) ----
#pragma unroll
        for (int ks = 0; ks < 2; ++ks) {
            const _Float16* pp = &sP[wave][lp][ks * 32];
            V16U aP;
            aP.h[0] = *(const h8*)(pp + kb);
            aP.h[1] = *(const h8*)(pp + 16 + kb);
#pragma unroll
            for (int dt = 0; dt < 4; ++dt) {
                const _Float16* vp = &sVT[dt * 16 + lp][ks * 32];
                V16U bV;
                bV.h[0] = *(const h8*)(vp + kb);
                bV.h[1] = *(const h8*)(vp + 16 + kb);
                acc[dt] = __builtin_amdgcn_wmma_f32_16x16x32_f16(false, aP.v, false, bV.v,
                                                                 (short)0, acc[dt],
                                                                 false, false);
            }
        }
    }

    // ---- epilogue: O = acc / rowsum ----
    float inv[8];
#pragma unroll
    for (int r = 0; r < 8; ++r) inv[r] = 1.0f / rowsum[r];
#pragma unroll
    for (int dt = 0; dt < 4; ++dt) {
#pragma unroll
        for (int r = 0; r < 8; ++r) {
            O[hoff + (size_t)(m0 + half * 8 + r) * DKDIM + dt * 16 + lp] =
                acc[dt][r] * inv[r];
        }
    }
}

extern "C" void kernel_launch(void* const* d_in, const int* in_sizes, int n_in,
                              void* d_out, int out_size, void* d_ws, size_t ws_size,
                              hipStream_t stream) {
    const float* q = (const float*)d_in[0];
    const float* k = (const float*)d_in[1];
    const float* v = (const float*)d_in[2];
    // d_in[3] is the causal mask (int32); causality is computed analytically.
    float* o = (float*)d_out;
    (void)in_sizes; (void)n_in; (void)out_size; (void)d_ws; (void)ws_size;

    dim3 grid(SEQ / BM, HEADS, 1);
    dim3 block(32 * NWAVE, 1, 1);
    fa_fwd_wmma_kernel<<<grid, block, 0, stream>>>(q, k, v, o);
}
